// Attention_71047349010757
// MI455X (gfx1250) — matrix operand order repaired and checked by reading, not compiled, not run on hardware
//
#include <hip/hip_runtime.h>

typedef __attribute__((ext_vector_type(16))) __bf16 v16bf;
typedef __attribute__((ext_vector_type(8)))  __bf16 bf16x8;
typedef __attribute__((ext_vector_type(8)))  float  v8f;

#define SEQ   1025
#define NP    1056          // padded seq (33*32)
#define CDIM  768
#define HD    64
#define NHEAD 12
#define NB    16
#define BHCNT (NB*NHEAD)    // 192
#define MROWS (NB*SEQ)      // 16400

static __device__ __forceinline__ __bf16 f2bf(float f) {
  unsigned u = __builtin_bit_cast(unsigned, f);
  u += 0x7FFFu + ((u >> 16) & 1u);                 // round-to-nearest-even
  unsigned short s = (unsigned short)(u >> 16);
  return __builtin_bit_cast(__bf16, s);
}

static __device__ __forceinline__ v8f zero8() {
  v8f z;
#pragma unroll
  for (int i = 0; i < 8; ++i) z[i] = 0.0f;
  return z;
}

// A-operand (16x32 bf16), row-major source: lane L -> row row0+(L%16);
// elems 0..7 = K kbase+8h..+7, elems 8..15 = K kbase+16+8h..+7 (h = L/16)
static __device__ __forceinline__ v16bf load_a(const __bf16* base, int ld, int row0, int kbase) {
  int lane = threadIdx.x & 31;
  int m = lane & 15, h = lane >> 4;
  const __bf16* p = base + (size_t)(row0 + m) * ld + (kbase + 8 * h);
  v16bf r;
  ((bf16x8*)&r)[0] = *(const bf16x8*)(p);
  ((bf16x8*)&r)[1] = *(const bf16x8*)(p + 16);
  return r;
}

// branch-free variant: clamps the row (padding rows read duplicated real data;
// their influence is masked downstream, so no zero-fill is required)
static __device__ __forceinline__ v16bf load_a_clamp(const __bf16* base, int ld, int row0,
                                                     int kbase, int nrows) {
  int lane = threadIdx.x & 31;
  int m = lane & 15, h = lane >> 4;
  int row = row0 + m;
  row = row < nrows ? row : (nrows - 1);
  const __bf16* p = base + (size_t)row * ld + (kbase + 8 * h);
  v16bf r;
  ((bf16x8*)&r)[0] = *(const bf16x8*)(p);
  ((bf16x8*)&r)[1] = *(const bf16x8*)(p + 16);
  return r;
}

// B-operand (32x16 bf16) from row-major B^T storage: lane L -> col col0+(L%16);
// elems 0..7 = K kbase+8h..+7, elems 8..15 = K kbase+16+8h..+7 (h = L/16)
static __device__ __forceinline__ v16bf load_b(const __bf16* baseT, int ld, int col0, int kbase) {
  int lane = threadIdx.x & 31;
  int n = lane & 15, h = lane >> 4;
  const __bf16* p = baseT + (size_t)(col0 + n) * ld + (kbase + 8 * h);
  v16bf r;
  ((bf16x8*)&r)[0] = *(const bf16x8*)(p);
  ((bf16x8*)&r)[1] = *(const bf16x8*)(p + 16);
  return r;
}

static __device__ __forceinline__ v8f wmma_bf16(v16bf a, v16bf b, v8f c) {
  return __builtin_amdgcn_wmma_f32_16x16x32_bf16(false, a, false, b, (short)0, c, false, false);
}

// ---------------- f32 -> bf16 conversion ----------------
__global__ void cvt_f32_bf16(const float* __restrict__ in, __bf16* __restrict__ out, int n) {
  int i = blockIdx.x * blockDim.x + threadIdx.x;
  if (i < n) out[i] = f2bf(in[i]);
}

// ---------------- GEMM1: qkv = x @ qkv_w^T, scatter to q,k,vT (bf16) ----------------
// 32x64 tile per wave. grid: x = 36 col groups, y = 132 (4 tiles/block, 33 tiles/batch)
// q is pre-scaled by 0.125 (= hd^-0.5) so attention needs no per-logit multiply.
__global__ void gemm_qkv(const __bf16* __restrict__ xb, const __bf16* __restrict__ wb,
                         __bf16* __restrict__ qo, __bf16* __restrict__ ko,
                         __bf16* __restrict__ vto) {
  int wid = threadIdx.x >> 5;
  int lane = threadIdx.x & 31;
  int t = blockIdx.y * 4 + wid;          // row tile 0..527
  int b = t / 33;
  int n0 = (t % 33) * 32;
  int col0 = blockIdx.x * 64;
  int s = col0 / CDIM;                   // 0=q 1=k 2=v
  int head = (col0 % CDIM) / HD;
  int m = lane & 15, h = lane >> 4;

  const __bf16* xrow = xb + (size_t)b * SEQ * CDIM;
  v8f acc[2][4];
#pragma unroll
  for (int u = 0; u < 2; ++u)
#pragma unroll
    for (int c = 0; c < 4; ++c) acc[u][c] = zero8();

  for (int kk = 0; kk < CDIM / 32; ++kk) {
    int kof = kk * 32;
    v16bf a0 = load_a_clamp(xrow, CDIM, n0, kof, SEQ);
    v16bf a1 = load_a_clamp(xrow, CDIM, n0 + 16, kof, SEQ);
#pragma unroll
    for (int c = 0; c < 4; ++c) {
      v16bf bm = load_b(wb, CDIM, col0 + 16 * c, kof);
      acc[0][c] = wmma_bf16(a0, bm, acc[0][c]);
      acc[1][c] = wmma_bf16(a1, bm, acc[1][c]);
    }
  }

  int bh = b * NHEAD + head;
#pragma unroll
  for (int u = 0; u < 2; ++u) {
#pragma unroll
    for (int c = 0; c < 4; ++c) {
#pragma unroll
      for (int r = 0; r < 8; ++r) {
        int n = n0 + 16 * u + r + 8 * h;
        int e = 16 * c + m;              // hd index within head
        float v = acc[u][c][r];
        if (s == 0)      qo[((size_t)bh * NP + n) * HD + e] = f2bf(v * 0.125f);
        else if (s == 1) ko[((size_t)bh * NP + n) * HD + e] = f2bf(v);
        else             vto[((size_t)bh * HD + e) * NP + n] = f2bf(v);  // transposed V
      }
    }
  }
}

// ---------------- fused attention: single-pass, no LDS, 32 queries/wave ----------------
// grid: x = 9 (4 query tiles of 32 per workgroup, 33 total), y = 192 (b*h)
__global__ void attn_fused(const __bf16* __restrict__ q, const __bf16* __restrict__ k,
                           const __bf16* __restrict__ vt, __bf16* __restrict__ xstd) {
  int wid = threadIdx.x >> 5;
  int lane = threadIdx.x & 31;
  int qt = blockIdx.x * 4 + wid;
  if (qt >= 33) return;                  // wave-uniform
  int bh = blockIdx.y;
  int b = bh / NHEAD, head = bh % NHEAD;
  int m = lane & 15, h = lane >> 4;

  const __bf16* qptr = q  + (size_t)bh * NP * HD;
  const __bf16* kptr = k  + (size_t)bh * NP * HD;
  const __bf16* vptr = vt + (size_t)bh * HD * NP;
  int q0 = qt * 32;

  // Q^T B-operands for 2 query sub-tiles x 2 K-chunks, resident all pass
  v16bf qB[2][2];
#pragma unroll
  for (int u = 0; u < 2; ++u) {
    qB[u][0] = load_b(qptr, HD, q0 + 16 * u, 0);
    qB[u][1] = load_b(qptr, HD, q0 + 16 * u, 32);
  }

  v8f o[2][4];
#pragma unroll
  for (int u = 0; u < 2; ++u)
#pragma unroll
    for (int c = 0; c < 4; ++c) o[u][c] = zero8();
  float sum[2] = {0.0f, 0.0f};

  for (int j = 0; j < NP / 32; ++j) {
    int kb0 = j * 32;
    // K A-operands shared by both query sub-tiles
    v16bf ka00 = load_a(kptr, HD, kb0,      0);
    v16bf ka01 = load_a(kptr, HD, kb0,      32);
    v16bf ka10 = load_a(kptr, HD, kb0 + 16, 0);
    v16bf ka11 = load_a(kptr, HD, kb0 + 16, 32);

    v16bf pp[2];
#pragma unroll
    for (int u = 0; u < 2; ++u) {
      // S^T tiles (keys x queries): lane = query, vgpr = key (q pre-scaled by 1/8)
      v8f s0 = zero8(), s1 = zero8();
      s0 = wmma_bf16(ka00, qB[u][0], s0);
      s0 = wmma_bf16(ka01, qB[u][1], s0);
      s1 = wmma_bf16(ka10, qB[u][0], s1);
      s1 = wmma_bf16(ka11, qB[u][1], s1);
      // exp (|logit| small: no max shift), key mask, lane-local repack to A-operand
#pragma unroll
      for (int r = 0; r < 8; ++r) {
        int key0 = kb0 + 8 * h + r;
        int key1 = kb0 + 16 + 8 * h + r;
        float e0 = (key0 < SEQ) ? __expf(s0[r]) : 0.0f;
        float e1 = (key1 < SEQ) ? __expf(s1[r]) : 0.0f;
        sum[u] += e0 + e1;
        pp[u][r]     = f2bf(e0);
        pp[u][r + 8] = f2bf(e1);
      }
    }

    // O += P(16x32) @ V(32x16); each V B-operand feeds both query sub-tiles
#pragma unroll
    for (int c = 0; c < 4; ++c) {
      v16bf vB = load_b(vptr, NP, c * 16, kb0);
      o[0][c] = wmma_bf16(pp[0], vB, o[0][c]);
      o[1][c] = wmma_bf16(pp[1], vB, o[1][c]);
    }
  }

#pragma unroll
  for (int u = 0; u < 2; ++u) {
    // softmax denominator per query (lane L holds query q0+16u+(L%16))
    float tot = sum[u] + __shfl_xor(sum[u], 16, 32);
    float inv = 1.0f / tot;
    float invr[8];
#pragma unroll
    for (int r = 0; r < 8; ++r) invr[r] = __shfl(inv, r + 8 * h, 32);  // query r+8h

#pragma unroll
    for (int c = 0; c < 4; ++c) {
#pragma unroll
      for (int r = 0; r < 8; ++r) {
        int n = q0 + 16 * u + r + 8 * h;
        if (n < SEQ)
          xstd[((size_t)(b * SEQ + n)) * CDIM + head * HD + c * 16 + m] =
              f2bf(o[u][c][r] * invr[r]);
      }
    }
  }
}

// ---------------- GEMM2: out = x_std @ proj_w^T + proj_b (f32 out) ----------------
// 32x64 tile per wave. grid: x = 12 col groups, y = 129 (4 tiles/block, 513 tiles)
__global__ void gemm_proj(const __bf16* __restrict__ xstd, const __bf16* __restrict__ pw,
                          const float* __restrict__ pb, float* __restrict__ out) {
  int wid = threadIdx.x >> 5;
  int lane = threadIdx.x & 31;
  int t = blockIdx.y * 4 + wid;
  if (t >= 513) return;                  // wave-uniform; 513*32 covers 16400 rows
  int row0 = t * 32;
  int col0 = blockIdx.x * 64;
  int m = lane & 15, h = lane >> 4;

  v8f acc[2][4];
#pragma unroll
  for (int u = 0; u < 2; ++u)
#pragma unroll
    for (int c = 0; c < 4; ++c) acc[u][c] = zero8();

  for (int kk = 0; kk < CDIM / 32; ++kk) {
    int kof = kk * 32;
    v16bf a0 = load_a_clamp(xstd, CDIM, row0, kof, MROWS);
    v16bf a1 = load_a_clamp(xstd, CDIM, row0 + 16, kof, MROWS);
#pragma unroll
    for (int c = 0; c < 4; ++c) {
      v16bf bm = load_b(pw, CDIM, col0 + 16 * c, kof);
      acc[0][c] = wmma_bf16(a0, bm, acc[0][c]);
      acc[1][c] = wmma_bf16(a1, bm, acc[1][c]);
    }
  }

#pragma unroll
  for (int u = 0; u < 2; ++u) {
#pragma unroll
    for (int c = 0; c < 4; ++c) {
      int col = col0 + 16 * c + m;
      float bias = pb[col];
#pragma unroll
      for (int r = 0; r < 8; ++r) {
        int row = row0 + 16 * u + r + 8 * h;
        if (row < MROWS)
          out[(size_t)row * CDIM + col] = acc[u][c][r] + bias;
      }
    }
  }
}

extern "C" void kernel_launch(void* const* d_in, const int* in_sizes, int n_in,
                              void* d_out, int out_size, void* d_ws, size_t ws_size,
                              hipStream_t stream) {
  const float* x      = (const float*)d_in[0];  // (16,1025,768)
  const float* qkv_w  = (const float*)d_in[1];  // (2304,768)
  const float* proj_w = (const float*)d_in[2];  // (768,768)
  const float* proj_b = (const float*)d_in[3];  // (768,)

  char* ws = (char*)d_ws;
  size_t off = 0;
  const size_t qsz  = (size_t)BHCNT * NP * HD * 2;   // 25.95 MB each
  const size_t xssz = (size_t)NB * SEQ * CDIM * 2;   // 25.19 MB each
  __bf16* qb    = (__bf16*)(ws + off); off += qsz;
  __bf16* kb    = (__bf16*)(ws + off); off += qsz;
  __bf16* vtb   = (__bf16*)(ws + off); off += qsz;
  __bf16* xstd  = (__bf16*)(ws + off); off += xssz;
  __bf16* x_bf  = (__bf16*)(ws + off); off += xssz;
  __bf16* qw_bf = (__bf16*)(ws + off); off += (size_t)3 * CDIM * CDIM * 2;
  __bf16* pw_bf = (__bf16*)(ws + off); off += (size_t)CDIM * CDIM * 2;

  int nx  = NB * SEQ * CDIM;         // 12,595,200
  int nqw = 3 * CDIM * CDIM;         // 1,769,472
  int npw = CDIM * CDIM;             // 589,824
  cvt_f32_bf16<<<(nx  + 255) / 256, 256, 0, stream>>>(x,      x_bf,  nx);
  cvt_f32_bf16<<<(nqw + 255) / 256, 256, 0, stream>>>(qkv_w,  qw_bf, nqw);
  cvt_f32_bf16<<<(npw + 255) / 256, 256, 0, stream>>>(proj_w, pw_bf, npw);

  gemm_qkv  <<<dim3(36, 132), 128, 0, stream>>>(x_bf, qw_bf, qb, kb, vtb);
  attn_fused<<<dim3(9, 192),  128, 0, stream>>>(qb, kb, vtb, xstd);
  gemm_proj <<<dim3(12, 129), 128, 0, stream>>>(xstd, pw_bf, proj_b, (float*)d_out);
}
